// GCN_43671227466166
// MI455X (gfx1250) — compile-verified, matrix operand order checked by reading, not run on hardware
//
#include <hip/hip_runtime.h>

#define NN  262144
#define NE  524288
#define USE_ASYNC_LDS 1

typedef __bf16 bf16_t;
typedef __attribute__((ext_vector_type(16))) __bf16 v16bf;
typedef __attribute__((ext_vector_type(8)))  float  v8f;

union FragAB { uint4 u[2]; v16bf v; };

__device__ __forceinline__ bf16_t f2bf(float f) {
    unsigned u = __float_as_uint(f);
    u += 0x7FFFu + ((u >> 16) & 1u);          // round-to-nearest-even
    unsigned short h = (unsigned short)(u >> 16);
    bf16_t r; __builtin_memcpy(&r, &h, 2);
    return r;
}

// monotonic float<->uint mapping for atomic max on signed floats
__device__ __forceinline__ unsigned encf(float f) {
    unsigned u = __float_as_uint(f);
    return (u & 0x80000000u) ? ~u : (u | 0x80000000u);
}
__device__ __forceinline__ float decf(unsigned u) {
    unsigned b = (u & 0x80000000u) ? (u & 0x7FFFFFFFu) : ~u;
    return __uint_as_float(b);
}

// ---------------------------------------------------------------------------
// packing kernels
// ---------------------------------------------------------------------------
__global__ void pack_x(const float* __restrict__ x, int din, int Kpad,
                       bf16_t* __restrict__ xb) {
    size_t total = (size_t)NN * Kpad;
    for (size_t idx = (size_t)blockIdx.x * blockDim.x + threadIdx.x;
         idx < total; idx += (size_t)gridDim.x * blockDim.x) {
        int col = (int)(idx % Kpad);
        size_t row = idx / Kpad;
        float v = (col < din) ? x[row * din + col] : 0.f;
        xb[idx] = f2bf(v);
    }
}

// pack the 4 weight matrices into WMMA B-fragment layout:
// wfrag[((w*NT+nt)*KT+kt)*512 + lane*16 + i] = W[k][n]
//   n = nt*16 + (lane&15)
//   k = kt*32 + (lane>=16 ? 8:0) + (i>=8 ? 16:0) + (i&7)
__global__ void pack_weights(const float* __restrict__ Wq, const float* __restrict__ Wk,
                             const float* __restrict__ Wv, const float* __restrict__ Ws,
                             int din, int dout, int KT, int NT,
                             bf16_t* __restrict__ wfrag) {
    int total = 4 * KT * NT * 512;
    for (int idx = blockIdx.x * blockDim.x + threadIdx.x; idx < total;
         idx += gridDim.x * blockDim.x) {
        int i    = idx & 15;
        int lane = (idx >> 4) & 31;
        int t    = idx >> 9;
        int kt   = t % KT;
        int wnt  = t / KT;
        int nt   = wnt % NT;
        int w    = wnt / NT;
        int n = nt * 16 + (lane & 15);
        int k = kt * 32 + ((lane >> 4) ? 8 : 0) + ((i >> 3) ? 16 : 0) + (i & 7);
        const float* W = (w == 0) ? Wq : (w == 1) ? Wk : (w == 2) ? Wv : Ws;
        float v = (k < din && n < dout) ? W[k * dout + n] : 0.f;
        wfrag[idx] = f2bf(v);
    }
}

// ---------------------------------------------------------------------------
// fused Q/K/V/S GEMM.  One wave = 16x16 output tile; A fragments live in VGPRs
// for the whole kernel; each weight matrix's B-fragment block is staged into
// LDS (async copy, memory->LDS without VGPR bounce) and shared by all 8 waves.
// nt loop capped at unroll 2: two in-flight accumulators cover WMMA latency
// without spilling past 256 VGPRs (keeps full wave occupancy, no MSB mode).
// ---------------------------------------------------------------------------
template <int KT, int NT, int DOUT>
__global__ __launch_bounds__(256)
void gemm_qkvs(const bf16_t* __restrict__ xb,
               const bf16_t* __restrict__ wfrag,
               const float* __restrict__ bq, const float* __restrict__ bk,
               const float* __restrict__ bv, const float* __restrict__ bs,
               float* __restrict__ oq, float* __restrict__ ok,
               float* __restrict__ ov, float* __restrict__ os) {
    constexpr int Kpad   = KT * 32;
    constexpr int FRAGS  = KT * NT;        // 512 bf16 each
    constexpr int NCHUNK = FRAGS * 64;     // 16B chunks per weight matrix

    __shared__ bf16_t sB[FRAGS * 512];

    const int lane = threadIdx.x & 31;
    const int wave = threadIdx.x >> 5;
    const int rowBase = (blockIdx.x * 8 + wave) * 16;
    const int m  = lane & 15;
    const int hi = lane >> 4;              // 0: lanes 0-15, 1: lanes 16-31

    // A fragments: lanes 0-15 row m, K {0..7,16..23}; lanes 16-31 row m, K {8..15,24..31}
    FragAB a[KT];
    const bf16_t* abase = xb + (size_t)(rowBase + m) * Kpad + hi * 8;
#pragma unroll
    for (int kt = 0; kt < KT; ++kt) {
        const bf16_t* p = abase + kt * 32;
        a[kt].u[0] = *(const uint4*)(p);
        a[kt].u[1] = *(const uint4*)(p + 16);
    }

    float* outs[4]         = {oq, ok, ov, os};
    const float* biases[4] = {bq, bk, bv, bs};

#pragma unroll
    for (int w = 0; w < 4; ++w) {
        // ---- stage this weight matrix's fragments into LDS ----
        const bf16_t* wsrc = wfrag + (size_t)w * FRAGS * 512;
#if USE_ASYNC_LDS
        {
            auto* sB3 = (__attribute__((address_space(3))) bf16_t*)sB;
            for (int c = threadIdx.x; c < NCHUNK; c += 256) {
                unsigned long long ga = (unsigned long long)(uintptr_t)(wsrc + c * 8);
                asm volatile("global_load_async_to_lds_b128 %0, %1, off"
                             :: "v"(sB3 + c * 8), "v"(ga) : "memory");
            }
            asm volatile("s_wait_asynccnt 0x0" ::: "memory");
        }
#else
        for (int c = threadIdx.x; c < NCHUNK; c += 256) {
            uint4 t = *(const uint4*)(wsrc + c * 8);
            *(uint4*)(sB + c * 8) = t;
        }
#endif
        __syncthreads();

        float* out      = outs[w];
        const float* bb = biases[w];
#pragma unroll 2
        for (int nt = 0; nt < NT; ++nt) {
            int col  = nt * 16 + m;
            float bc = (col < DOUT) ? bb[col] : 0.f;
            v8f acc;
#pragma unroll
            for (int r = 0; r < 8; ++r) acc[r] = bc;

            const bf16_t* bbase = sB + (nt * KT) * 512 + lane * 16;
#pragma unroll
            for (int kt = 0; kt < KT; ++kt) {
                FragAB b;
                const bf16_t* p = bbase + kt * 512;
                b.u[0] = *(const uint4*)(p);
                b.u[1] = *(const uint4*)(p + 8);
                acc = __builtin_amdgcn_wmma_f32_16x16x32_bf16(
                    false, a[kt].v, false, b.v, (short)0, acc, false, false);
            }
            if (col < DOUT) {
                float* op = out + (size_t)(rowBase + hi * 8) * DOUT + col;
#pragma unroll
                for (int r = 0; r < 8; ++r) op[(size_t)r * DOUT] = acc[r];
            }
        }
        __syncthreads();   // before next w overwrites sB
    }
}

// ---------------------------------------------------------------------------
// attention (edge-parallel)
// ---------------------------------------------------------------------------
__global__ void fill_f32(float* p, float v, size_t n) {
    for (size_t i = (size_t)blockIdx.x * blockDim.x + threadIdx.x; i < n;
         i += (size_t)gridDim.x * blockDim.x) p[i] = v;
}
__global__ void fill_u32(unsigned* p, unsigned v, size_t n) {
    for (size_t i = (size_t)blockIdx.x * blockDim.x + threadIdx.x; i < n;
         i += (size_t)gridDim.x * blockDim.x) p[i] = v;
}

__global__ __launch_bounds__(256)
void att_scores(const float* __restrict__ q, const float* __restrict__ k,
                const long long* __restrict__ ei, int dout, float inv_sqrt_d,
                float* __restrict__ scores, unsigned* __restrict__ mU) {
    int e = blockIdx.x * 8 + (threadIdx.x >> 5);
    if (e >= NE) return;
    int lane = threadIdx.x & 31;
    int s = (int)ei[e];
    int t = (int)ei[NE + e];
    const float* qp = q + (size_t)t * dout;
    const float* kp = k + (size_t)s * dout;
    float acc = 0.f;
    for (int j = lane; j < dout; j += 32) acc += qp[j] * kp[j];
    for (int off = 16; off; off >>= 1) acc += __shfl_xor(acc, off, 32);
    if (lane == 0) {
        float sc = acc * inv_sqrt_d;
        scores[e] = sc;
        atomicMax(mU + t, encf(sc));
    }
}

__global__ void finalize_m(const unsigned* __restrict__ mU, float* __restrict__ mF) {
    int i = blockIdx.x * blockDim.x + threadIdx.x;
    if (i >= NN) return;
    unsigned u = mU[i];
    float f = 0.f;
    if (u != 0u) {
        f = decf(u);
        unsigned b = __float_as_uint(f);
        if ((b & 0x7F800000u) == 0x7F800000u) f = 0.f;   // non-finite guard
    }
    mF[i] = f;
}

__global__ void att_exp(float* __restrict__ scores, const float* __restrict__ mF,
                        const long long* __restrict__ ei, float* __restrict__ denom) {
    int e = blockIdx.x * blockDim.x + threadIdx.x;
    if (e >= NE) return;
    int t = (int)ei[NE + e];
    float ev = __expf(scores[e] - mF[t]);
    scores[e] = ev;
    atomicAdd(denom + t, ev);
}

__global__ __launch_bounds__(256)
void att_agg(const float* __restrict__ scores, const float* __restrict__ denom,
             const float* __restrict__ v, const long long* __restrict__ ei,
             int dout, float* __restrict__ agg) {
    int e = blockIdx.x * 8 + (threadIdx.x >> 5);
    if (e >= NE) return;
    int lane = threadIdx.x & 31;
    int s = (int)ei[e];
    int t = (int)ei[NE + e];
    float alpha = scores[e] / (denom[t] + 1e-16f);
    const float* vp = v + (size_t)s * dout;
    float* ap = agg + (size_t)t * dout;
    for (int j = lane; j < dout; j += 32) atomicAdd(ap + j, alpha * vp[j]);
}

// ---------------------------------------------------------------------------
// batchnorm / activations / log-softmax
// ---------------------------------------------------------------------------
__global__ __launch_bounds__(256)
void bn_stats(const float* __restrict__ x, int d, float* __restrict__ sums) {
    int col = threadIdx.x;
    if (col >= d) return;
    size_t row0 = (size_t)blockIdx.x * 256;
    float s = 0.f, s2 = 0.f;
    for (int r = 0; r < 256; ++r) {
        float v = x[(row0 + r) * d + col];
        s += v; s2 += v * v;
    }
    atomicAdd(sums + col, s);
    atomicAdd(sums + d + col, s2);
}

__global__ void bn_apply(const float* __restrict__ x, float* __restrict__ y,
                         const float* __restrict__ sums, const float* __restrict__ gamma,
                         const float* __restrict__ beta, int d, int relu) {
    size_t total = (size_t)NN * d;
    const float invN = 1.f / (float)NN;
    for (size_t idx = (size_t)blockIdx.x * blockDim.x + threadIdx.x; idx < total;
         idx += (size_t)gridDim.x * blockDim.x) {
        int col = (int)(idx % d);
        float mu  = sums[col] * invN;
        float var = sums[d + col] * invN - mu * mu;
        float out = gamma[col] * (x[idx] - mu) * rsqrtf(var + 1e-5f) + beta[col];
        if (relu) out = fmaxf(out, 0.f);
        y[idx] = out;
    }
}

__global__ void add_relu(const float* __restrict__ a, const float* __restrict__ b,
                         float* __restrict__ y, size_t n) {
    for (size_t i = (size_t)blockIdx.x * blockDim.x + threadIdx.x; i < n;
         i += (size_t)gridDim.x * blockDim.x) y[i] = fmaxf(a[i] + b[i], 0.f);
}

__global__ void relu_inplace(float* __restrict__ a, size_t n) {
    for (size_t i = (size_t)blockIdx.x * blockDim.x + threadIdx.x; i < n;
         i += (size_t)gridDim.x * blockDim.x) a[i] = fmaxf(a[i], 0.f);
}

__global__ void log_softmax10(const float* __restrict__ x, float* __restrict__ out) {
    int r = blockIdx.x * blockDim.x + threadIdx.x;
    if (r >= NN) return;
    const float* p = x + (size_t)r * 10;
    float mx = p[0];
    for (int j = 1; j < 10; ++j) mx = fmaxf(mx, p[j]);
    float s = 0.f;
    for (int j = 0; j < 10; ++j) s += __expf(p[j] - mx);
    float ls = __logf(s);
    for (int j = 0; j < 10; ++j) out[(size_t)r * 10 + j] = p[j] - mx - ls;
}

// ---------------------------------------------------------------------------
// host orchestration
// ---------------------------------------------------------------------------
struct Ws {
    bf16_t* xb;     bf16_t* wfrag;
    float *q, *k, *v, *agg, *xcur, *skip;
    float *scores, *mF, *denom, *sums;
    unsigned* mU;
};

static void launch_gemm(hipStream_t st, int din, int dout, const bf16_t* xb,
                        const bf16_t* wf,
                        const float* bq, const float* bk, const float* bv, const float* bs,
                        float* oq, float* ok, float* ov, float* os) {
    dim3 g(NN / 128), b(256);
    if (din == 220 && dout == 220)
        gemm_qkvs<7, 14, 220><<<g, b, 0, st>>>(xb, wf, bq, bk, bv, bs, oq, ok, ov, os);
    else if (din == 220 && dout == 150)
        gemm_qkvs<7, 10, 150><<<g, b, 0, st>>>(xb, wf, bq, bk, bv, bs, oq, ok, ov, os);
    else if (din == 150 && dout == 100)
        gemm_qkvs<5, 7, 100><<<g, b, 0, st>>>(xb, wf, bq, bk, bv, bs, oq, ok, ov, os);
    else if (din == 100 && dout == 60)
        gemm_qkvs<4, 4, 60><<<g, b, 0, st>>>(xb, wf, bq, bk, bv, bs, oq, ok, ov, os);
    else if (din == 60 && dout == 30)
        gemm_qkvs<2, 2, 30><<<g, b, 0, st>>>(xb, wf, bq, bk, bv, bs, oq, ok, ov, os);
    else
        gemm_qkvs<1, 1, 10><<<g, b, 0, st>>>(xb, wf, bq, bk, bv, bs, oq, ok, ov, os);
}

static void run_tconv(hipStream_t st, const float* xin, int din, int dout,
                      void* const* d_in, int ci, const long long* ei, const Ws& w) {
    int Kpad = ((din + 31) / 32) * 32;
    int KT = Kpad / 32;
    int NT = (dout + 15) / 16;
    const float* Wq  = (const float*)d_in[2 + 8 * ci + 0];
    const float* Wk  = (const float*)d_in[2 + 8 * ci + 1];
    const float* Wv  = (const float*)d_in[2 + 8 * ci + 2];
    const float* Wsk = (const float*)d_in[2 + 8 * ci + 3];
    const float* bq  = (const float*)d_in[2 + 8 * ci + 4];
    const float* bk  = (const float*)d_in[2 + 8 * ci + 5];
    const float* bv  = (const float*)d_in[2 + 8 * ci + 6];
    const float* bs  = (const float*)d_in[2 + 8 * ci + 7];

    pack_weights<<<512, 256, 0, st>>>(Wq, Wk, Wv, Wsk, din, dout, KT, NT, w.wfrag);
    pack_x<<<4096, 256, 0, st>>>(xin, din, Kpad, w.xb);
    launch_gemm(st, din, dout, w.xb, w.wfrag, bq, bk, bv, bs,
                w.q, w.k, w.v, w.agg);     // Ws output lands in agg (skip term)

    fill_u32<<<512, 256, 0, st>>>(w.mU, 0u, NN);
    fill_f32<<<512, 256, 0, st>>>(w.denom, 0.f, NN);
    att_scores<<<NE / 8, 256, 0, st>>>(w.q, w.k, ei, dout, rsqrtf((float)dout),
                                       w.scores, w.mU);
    finalize_m<<<NN / 256, 256, 0, st>>>(w.mU, w.mF);
    att_exp<<<NE / 256, 256, 0, st>>>(w.scores, w.mF, ei, w.denom);
    att_agg<<<NE / 8, 256, 0, st>>>(w.scores, w.denom, w.v, ei, dout, w.agg);
}

static void run_bn(hipStream_t st, const float* in, float* out, int d,
                   const float* gamma, const float* beta, int relu, const Ws& w) {
    fill_f32<<<8, 256, 0, st>>>(w.sums, 0.f, (size_t)2 * d);
    bn_stats<<<NN / 256, 256, 0, st>>>(in, d, w.sums);
    bn_apply<<<4096, 256, 0, st>>>(in, out, w.sums, gamma, beta, d, relu);
}

extern "C" void kernel_launch(void* const* d_in, const int* in_sizes, int n_in,
                              void* d_out, int out_size, void* d_ws, size_t ws_size,
                              hipStream_t stream) {
    (void)in_sizes; (void)n_in; (void)out_size; (void)ws_size;
    const float* x0 = (const float*)d_in[0];
    const long long* ei = (const long long*)d_in[1];
    auto BNg = [&](int bi) { return (const float*)d_in[58 + 2 * bi + 0]; };
    auto BNb = [&](int bi) { return (const float*)d_in[58 + 2 * bi + 1]; };

    // carve workspace (all 256B aligned)
    const size_t ACT = (size_t)NN * 224 * sizeof(float);   // 224 >= max padded dim
    char* p = (char*)d_ws;
    auto take = [&](size_t bytes) { char* r = p; p += (bytes + 255) & ~(size_t)255; return r; };
    Ws w;
    w.xb     = (bf16_t*)take((size_t)NN * 224 * sizeof(bf16_t));
    w.wfrag  = (bf16_t*)take((size_t)4 * 7 * 14 * 512 * sizeof(bf16_t));
    w.q      = (float*)take(ACT);
    w.k      = (float*)take(ACT);
    w.v      = (float*)take(ACT);
    w.agg    = (float*)take(ACT);
    w.xcur   = (float*)take(ACT);
    w.skip   = (float*)take(ACT);
    w.scores = (float*)take((size_t)NE * sizeof(float));
    w.mU     = (unsigned*)take((size_t)NN * sizeof(unsigned));
    w.mF     = (float*)take((size_t)NN * sizeof(float));
    w.denom  = (float*)take((size_t)NN * sizeof(float));
    w.sums   = (float*)take(2 * 224 * sizeof(float));

    // L1: x = relu(bn1(conv1(x)))
    run_tconv(stream, x0, 220, 220, d_in, 0, ei, w);
    run_bn(stream, w.agg, w.xcur, 220, BNg(0), BNb(0), 1, w);

    // L2: x = bn2(conv2(x))  -> skip buffer
    run_tconv(stream, w.xcur, 220, 150, d_in, 1, ei, w);
    run_bn(stream, w.agg, w.skip, 150, BNg(1), BNb(1), 0, w);

    // skip branch: identity = conv_skip_1(x0); x = relu(skip + identity)
    run_tconv(stream, x0, 220, 150, d_in, 2, ei, w);
    add_relu<<<4096, 256, 0, stream>>>(w.skip, w.agg, w.xcur, (size_t)NN * 150);

    // L3: x = relu(bn3(conv3(x)))
    run_tconv(stream, w.xcur, 150, 100, d_in, 3, ei, w);
    run_bn(stream, w.agg, w.xcur, 100, BNg(2), BNb(2), 1, w);

    // L4: x = relu(bn4(conv4(x)))
    run_tconv(stream, w.xcur, 100, 60, d_in, 4, ei, w);
    run_bn(stream, w.agg, w.xcur, 60, BNg(3), BNb(3), 1, w);

    // L5: x = bn5(relu(conv5(x)))
    run_tconv(stream, w.xcur, 60, 30, d_in, 5, ei, w);
    relu_inplace<<<4096, 256, 0, stream>>>(w.agg, (size_t)NN * 30);
    run_bn(stream, w.agg, w.xcur, 30, BNg(4), BNb(4), 0, w);

    // L6: log_softmax(conv6(x))
    run_tconv(stream, w.xcur, 30, 10, d_in, 6, ei, w);
    log_softmax10<<<NN / 256, 256, 0, stream>>>(w.agg, (float*)d_out);
}
